// TimesNet_25237227831546
// MI455X (gfx1250) — matrix operand order repaired and checked by reading, compiled, not run on hardware
//
#include <hip/hip_runtime.h>
#include <hip/hip_bf16.h>

typedef __attribute__((ext_vector_type(16))) _Float16 v16h;
typedef __attribute__((ext_vector_type(8)))  _Float16 v8h;
typedef __attribute__((ext_vector_type(8)))  float    v8f;

#define B_   32
#define T_   512
#define N_   64
#define D_   256
#define LMAX 1536   // 3 * 512 worst-case folded length
#define PRED 96

// ---------------------------------------------------------------- init header
__global__ void k_init(int* hdr) { hdr[0] = 0; hdr[1] = 0; }

// ------------------------------------------------- periods: DFT mag^2 + top-3
__global__ __launch_bounds__(256) void k_periods(const float* __restrict__ x,
                                                 int* hdr, int* __restrict__ Pout) {
  __shared__ float xs[T_];
  __shared__ float mag[257];
  const int n = blockIdx.x, b = blockIdx.y, tid = threadIdx.x;
  xs[tid]       = x[((size_t)b * T_ + tid)       * N_ + n];
  xs[tid + 256] = x[((size_t)b * T_ + tid + 256) * N_ + n];
  __syncthreads();
  // thread tid handles frequency bin f = tid+1 (bins 1..256); bin 0 forced 0
  const int f = tid + 1;
  float cr = 0.f, ci = 0.f;
  for (int t = 0; t < T_; ++t) {
    int ph = (f * t) & (T_ - 1);                 // exact phase mod 512
    float ang = (float)ph * (6.283185307179586f / (float)T_);
    float s, c;
    __sincosf(ang, &s, &c);
    cr += xs[t] * c;
    ci += xs[t] * s;
  }
  mag[f] = cr * cr + ci * ci;
  if (tid == 0) mag[0] = 0.f;
  __syncthreads();
  if (tid == 0) {
    int pmax = 0;
    for (int k = 0; k < 3; ++k) {
      float best = -1.f; int bi = 0;
      for (int q = 0; q < 257; ++q)
        if (mag[q] > best) { best = mag[q]; bi = q; }   // ties -> lowest index
      mag[bi] = -2.f;
      int fr = bi < 1 ? 1 : bi;
      int P = T_ / fr; if (P < 1) P = 1;
      Pout[(b * N_ + n) * 3 + k] = P;
      if (P > pmax) pmax = P;
    }
    atomicMax(hdr, pmax);
  }
}

__global__ void k_finalize(int* hdr) {
  int pm = hdr[0];
  int L = 3 * pm;
  if (L > LMAX) L = LMAX;
  hdr[1] = L;
}

// ------------------------------- fold periods -> f16, transposed [b][l][64ch]
__global__ __launch_bounds__(512) void k_fold(const float* __restrict__ x,
                                              const int* __restrict__ Pv,
                                              const int* __restrict__ hdr,
                                              _Float16* __restrict__ z016T) {
  const int k = blockIdx.x, n = blockIdx.y, b = blockIdx.z;
  const int p = threadIdx.x;
  const int Pmax = hdr[0];
  if (p >= Pmax) return;
  const int P = Pv[(b * N_ + n) * 3 + k];
  float m = 0.f;
  if (p < P) {
    int cyc = T_ / P; if (cyc < 1) cyc = 1;
    int start = T_ - cyc * P;
    for (int c = 0; c < cyc; ++c)
      m += x[((size_t)b * T_ + start + c * P + p) * N_ + n];
    m /= (float)cyc;
  }
  z016T[((size_t)b * LMAX + k * Pmax + p) * N_ + n] = (_Float16)m;
}

// ------------------------------------------------------------- weight prep
__global__ void k_cvt_stem(const float* __restrict__ w, _Float16* __restrict__ w16) {
  int i = blockIdx.x * 256 + threadIdx.x;
  if (i < D_ * N_) w16[i] = (_Float16)w[i];
}

// Effective k=7 conv weights: Weff[o][t][i] = sum_j sum_c Wp[o, j*256+c] * Wj[c,i,tl]
__global__ __launch_bounds__(256) void k_weff(const float* __restrict__ w0,
                                              const float* __restrict__ w1,
                                              const float* __restrict__ w2,
                                              const float* __restrict__ projw,
                                              _Float16* __restrict__ weff) {
  const int o = threadIdx.x, i = blockIdx.x, t = blockIdx.y;
  const int ks[3] = {3, 5, 7};
  const float* ws[3] = {w0, w1, w2};
  float acc = 0.f;
  for (int j = 0; j < 3; ++j) {
    int kj = ks[j], pad = kj >> 1;
    int tl = t - 3 + pad;
    if (tl < 0 || tl >= kj) continue;
    const float* wj = ws[j];
    const float* wp = projw + (size_t)o * (3 * D_) + j * D_;
    for (int c = 0; c < D_; ++c)
      acc += wp[c] * wj[((size_t)c * D_ + i) * kj + tl];
  }
  weff[((size_t)o * 7 + t) * D_ + i] = (_Float16)acc;
}

__global__ void k_beff(const float* __restrict__ b0, const float* __restrict__ b1,
                       const float* __restrict__ b2, const float* __restrict__ projw,
                       const float* __restrict__ projb, float* __restrict__ beff) {
  const int o = threadIdx.x;
  const float* bs[3] = {b0, b1, b2};
  float acc = projb[o];
  for (int j = 0; j < 3; ++j) {
    const float* wp = projw + (size_t)o * (3 * D_) + j * D_;
    for (int c = 0; c < D_; ++c) acc += wp[c] * bs[j][c];
  }
  beff[o] = acc;
}

// ------------------------------------------------------- stem: 64->256 GEMM
// z016T: [b][l][64ch] f16 ; out32: [b][ch][l] f32 ; out16T: [b][l][256ch] f16
__global__ __launch_bounds__(256) void k_stem(const _Float16* __restrict__ z016T,
                                              const _Float16* __restrict__ w16,
                                              const float* __restrict__ bias,
                                              const int* __restrict__ hdr,
                                              float* __restrict__ out32,
                                              _Float16* __restrict__ out16T) {
  const int L = hdr[1];
  const int l0 = blockIdx.x * 16;
  if (l0 >= L) return;
  const int b = blockIdx.z;
  const int lane = threadIdx.x & 31, wv = threadIdx.x >> 5;
  const int h = lane >> 4, n = lane & 15, mrow = lane & 15;
  const int m0 = blockIdx.y * 128 + wv * 16;
  const int col = l0 + n;
  const bool ok = col < L;

  v8f acc = {};
#pragma unroll
  for (int kc = 0; kc < 2; ++kc) {
    v8h blo = {}, bhi = {};
    if (ok) {
      const _Float16* bp = z016T + ((size_t)b * LMAX + col) * N_ + kc * 32 + 16 * h;
      blo = *(const v8h*)bp;
      bhi = *(const v8h*)(bp + 8);
    }
    v16h bb = __builtin_shufflevector(blo, bhi, 0, 1, 2, 3, 4, 5, 6, 7,
                                      8, 9, 10, 11, 12, 13, 14, 15);
    const _Float16* abase = w16 + (size_t)(m0 + mrow) * N_ + kc * 32;
    v8h alo = *(const v8h*)(abase + 16 * h);
    v8h ahi = *(const v8h*)(abase + 16 * h + 8);
    v16h a = __builtin_shufflevector(alo, ahi, 0, 1, 2, 3, 4, 5, 6, 7,
                                     8, 9, 10, 11, 12, 13, 14, 15);
    acc = __builtin_amdgcn_wmma_f32_16x16x32_f16(false, a, false, bb,
                                                 (short)0, acc, false, false);
  }
  if (ok) {
    v8h pack;
#pragma unroll
    for (int v = 0; v < 8; ++v) {
      int o = m0 + 8 * h + v;                       // contiguous over v
      float val = acc[v] + bias[o];
      out32[((size_t)(b * D_ + o)) * LMAX + col] = val;
      pack[v] = (_Float16)val;
    }
    *(v8h*)&out16T[((size_t)b * LMAX + col) * D_ + m0 + 8 * h] = pack;
  }
}

// ------------------------- inception block as single k=7 conv + GELU + resid
// in16T/out16T: [b][l][256ch] f16 ; in32/out32: [b][ch][l] f32
__global__ __launch_bounds__(256) void k_conv(const _Float16* __restrict__ in16T,
                                              const float* __restrict__ in32,
                                              const _Float16* __restrict__ weff,
                                              const float* __restrict__ beff,
                                              const int* __restrict__ hdr,
                                              float* __restrict__ out32,
                                              _Float16* __restrict__ out16T) {
  const int L = hdr[1];
  const int l0 = blockIdx.x * 16;
  if (l0 >= L) return;
  const int b = blockIdx.z;
  const int tid = threadIdx.x;
  const int lane = tid & 31, wv = tid >> 5;
  const int h = lane >> 4, n = lane & 15, mrow = lane & 15;
  const int m0 = blockIdx.y * 128 + wv * 16;

  // halo tile: 22 columns x 256 channels, channel-contiguous rows.
  // Row stride 264 halves (=132 dwords = 4 banks) -> b128 reads by the 16
  // column-lanes fan across all 64 LDS banks.
  __shared__ _Float16 Bext[22][264];

  // stage once: 22*32 16-byte chunks, vector loads from transposed input
  for (int e = tid; e < 22 * 32; e += 256) {
    int cc = e >> 5, ch8 = e & 31;
    int gc = l0 - 3 + cc;
    v8h v = {};
    if (gc >= 0 && gc < L)
      v = *(const v8h*)(in16T + ((size_t)b * LMAX + gc) * D_ + ch8 * 8);
    *(v8h*)&Bext[cc][ch8 * 8] = v;
  }
  __syncthreads();

  v8f acc = {};
  for (int kc = 0; kc < 8; ++kc) {
#pragma unroll
    for (int t = 0; t < 7; ++t) {
      const _Float16* abase = weff + ((size_t)(m0 + mrow) * 7 + t) * D_ + kc * 32;
      v8h alo = *(const v8h*)(abase + 16 * h);
      v8h ahi = *(const v8h*)(abase + 16 * h + 8);
      v16h a = __builtin_shufflevector(alo, ahi, 0, 1, 2, 3, 4, 5, 6, 7,
                                       8, 9, 10, 11, 12, 13, 14, 15);
      const _Float16* bp = &Bext[n + t][kc * 32 + 16 * h];
      v8h blo = *(const v8h*)bp;
      v8h bhi = *(const v8h*)(bp + 8);
      v16h bb = __builtin_shufflevector(blo, bhi, 0, 1, 2, 3, 4, 5, 6, 7,
                                        8, 9, 10, 11, 12, 13, 14, 15);
      acc = __builtin_amdgcn_wmma_f32_16x16x32_f16(false, a, false, bb,
                                                   (short)0, acc, false, false);
    }
  }

  const int col = l0 + n;
  if (col < L) {
    v8h pack;
#pragma unroll
    for (int v = 0; v < 8; ++v) {
      int o = m0 + 8 * h + v;                       // contiguous over v
      float val = acc[v] + beff[o];
      float ge = 0.5f * val * (1.0f + erff(val * 0.7071067811865475f)); // exact GELU
      size_t idx = ((size_t)(b * D_ + o)) * LMAX + col;
      float z = ge + in32[idx];
      out32[idx] = z;
      pack[v] = (_Float16)z;
    }
    *(v8h*)&out16T[((size_t)b * LMAX + col) * D_ + m0 + 8 * h] = pack;
  }
}

// ----------------------------------------- mean pool + head + broadcast to N
__global__ __launch_bounds__(256) void k_head(const float* __restrict__ z32,
                                              const float* __restrict__ head_w,
                                              const float* __restrict__ head_b,
                                              const int* __restrict__ hdr,
                                              float* __restrict__ out) {
  __shared__ float gs[D_];
  const int b = blockIdx.x, c = threadIdx.x;
  const int L = hdr[1];
  float s = 0.f;
  const float* row = z32 + ((size_t)(b * D_ + c)) * LMAX;
  for (int l = 0; l < L; ++l) s += row[l];
  gs[c] = s / (float)L;
  __syncthreads();
  if (c < PRED) {
    float acc = head_b[c];
    const float* hw = head_w + (size_t)c * D_;
    for (int q = 0; q < D_; ++q) acc += hw[q] * gs[q];
    float* o = out + ((size_t)b * PRED + c) * N_;
    for (int n = 0; n < N_; ++n) o[n] = acc;
  }
}

// ---------------------------------------------------------------------------
extern "C" void kernel_launch(void* const* d_in, const int* in_sizes, int n_in,
                              void* d_out, int out_size, void* d_ws, size_t ws_size,
                              hipStream_t stream) {
  (void)in_sizes; (void)n_in; (void)out_size; (void)ws_size;
  const float* x      = (const float*)d_in[0];
  const float* stem_w = (const float*)d_in[1];
  const float* stem_b = (const float*)d_in[2];
  const float* head_w = (const float*)d_in[3];
  const float* head_b = (const float*)d_in[4];
  const float* blk[2][8];
  int base = 5;
  for (int bl = 0; bl < 2; ++bl) {
    for (int q = 0; q < 8; ++q) blk[bl][q] = (const float*)d_in[base + q];
    base += 8;
  }
  // blk[bl]: 0=p0w 1=p0b 2=p1w 3=p1b 4=p2w 5=p2b 6=projw 7=projb

  char* w = (char*)d_ws;
  size_t off = 0;
  auto alloc = [&](size_t bytes) -> void* {
    void* p = w + off;
    off = (off + bytes + 255) & ~(size_t)255;
    return p;
  };
  int*      hdr     = (int*)alloc(256);
  int*      Pv      = (int*)alloc((size_t)B_ * N_ * 3 * sizeof(int));
  _Float16* stemW16 = (_Float16*)alloc((size_t)D_ * N_ * sizeof(_Float16));
  _Float16* weff    = (_Float16*)alloc((size_t)2 * D_ * 7 * D_ * sizeof(_Float16));
  float*    beff    = (float*)alloc((size_t)2 * D_ * sizeof(float));
  _Float16* z016T   = (_Float16*)alloc((size_t)B_ * LMAX * N_ * sizeof(_Float16));
  float*    zA32    = (float*)alloc((size_t)B_ * D_ * LMAX * sizeof(float));
  _Float16* zA16T   = (_Float16*)alloc((size_t)B_ * LMAX * D_ * sizeof(_Float16));
  float*    zB32    = (float*)alloc((size_t)B_ * D_ * LMAX * sizeof(float));
  _Float16* zB16T   = (_Float16*)alloc((size_t)B_ * LMAX * D_ * sizeof(_Float16));

  k_init<<<1, 1, 0, stream>>>(hdr);
  k_periods<<<dim3(N_, B_), 256, 0, stream>>>(x, hdr, Pv);
  k_finalize<<<1, 1, 0, stream>>>(hdr);
  k_fold<<<dim3(3, N_, B_), 512, 0, stream>>>(x, Pv, hdr, z016T);

  k_cvt_stem<<<(D_ * N_ + 255) / 256, 256, 0, stream>>>(stem_w, stemW16);
  for (int bl = 0; bl < 2; ++bl) {
    k_weff<<<dim3(D_, 7), 256, 0, stream>>>(blk[bl][0], blk[bl][2], blk[bl][4],
                                            blk[bl][6], weff + (size_t)bl * D_ * 7 * D_);
    k_beff<<<1, 256, 0, stream>>>(blk[bl][1], blk[bl][3], blk[bl][5],
                                  blk[bl][6], blk[bl][7], beff + bl * D_);
  }

  dim3 gemm_grid(LMAX / 16, 2, B_);  // 96 l-tiles x 2 m-tiles(128) x batch
  k_stem<<<gemm_grid, 256, 0, stream>>>(z016T, stemW16, stem_b, hdr, zA32, zA16T);
  k_conv<<<gemm_grid, 256, 0, stream>>>(zA16T, zA32, weff, beff, hdr, zB32, zB16T);
  k_conv<<<gemm_grid, 256, 0, stream>>>(zB16T, zB32, weff + (size_t)D_ * 7 * D_,
                                        beff + D_, hdr, zA32, zA16T);

  k_head<<<B_, 256, 0, stream>>>(zA32, head_w, head_b, hdr, (float*)d_out);
}
